// AttentionModule_45475113730220
// MI455X (gfx1250) — matrix operand order repaired and checked by reading, not compiled, not run on hardware
//
#include <hip/hip_runtime.h>
#include <hip/hip_bf16.h>
#include <stdint.h>

// Problem constants (match reference)
#define BB   4
#define CC   512
#define NN   4096
#define CQK  64
#define NT1  64     // n-slab per workgroup in projection kernel
#define QT   32     // queries per workgroup in attention kernel
#define OTOT 640    // total projection output rows: 64 q + 64 k + 512 v

typedef __attribute__((ext_vector_type(16))) _Float16 v16h;
typedef __attribute__((ext_vector_type(8)))  _Float16 v8h;
typedef __attribute__((ext_vector_type(8)))  float    v8f;
typedef __attribute__((ext_vector_type(4)))  int      v4i;

__device__ __forceinline__ v16h ld2x8h(const _Float16* p0, const _Float16* p1) {
    union { v16h v; v8h h[2]; } u;
    u.h[0] = *(const v8h*)p0;
    u.h[1] = *(const v8h*)p1;
    return u.v;
}

// --------------------------- async-to-LDS plumbing -------------------------
#if defined(__has_builtin)
#  if __has_builtin(__builtin_amdgcn_global_load_async_to_lds_b128)
#    define ASYNC_LDS 1
#  else
#    define ASYNC_LDS 0
#  endif
#else
#  define ASYNC_LDS 0
#endif

#if ASYNC_LDS
#  if __has_builtin(__builtin_amdgcn_s_wait_asynccnt)
#    define WAIT_ASYNC(n) __builtin_amdgcn_s_wait_asynccnt(n)
#  else
#    define WAIT_ASYNC(n) asm volatile("s_wait_asynccnt %0" :: "n"(n) : "memory")
#  endif
#else
#  define WAIT_ASYNC(n) ((void)0)
#endif

// Copy 16 bytes global -> LDS. Async (ASYNCcnt-tracked) when available.
__device__ __forceinline__ void stage16(_Float16* lds_dst, const _Float16* gsrc) {
#if ASYNC_LDS
    __builtin_amdgcn_global_load_async_to_lds_b128(
        (__attribute__((address_space(1))) v4i*)(uintptr_t)gsrc,
        (__attribute__((address_space(3))) v4i*)(uint32_t)(uintptr_t)lds_dst,
        0, 0);
#else
    *(v8h*)lds_dst = *(const v8h*)gsrc;
#endif
}

// ---------------------------------------------------------------------------
// Kernel 0: one-time f32 -> f16 conversion of [Wq;Wk;Wv] into Wh (640x512).
//   Removes all weight conversions from the projection GEMM hot loop.
// ---------------------------------------------------------------------------
__global__ __launch_bounds__(256) void convert_w_kernel(
    const float* __restrict__ Wq, const float* __restrict__ Wk,
    const float* __restrict__ Wv, _Float16* __restrict__ Wh)
{
    int i = blockIdx.x * 256 + threadIdx.x;     // 0 .. 640*512-1
    float v;
    if      (i <     CQK * CC) v = Wq[i];
    else if (i < 2 * CQK * CC) v = Wk[i - CQK * CC];
    else                       v = Wv[i - 2 * CQK * CC];
    Wh[i] = (_Float16)v;
}

// ---------------------------------------------------------------------------
// Kernel 1: fused QKV projection.
//   Wh (640x512 f16) as A-matrix, x[b] as 512xN B-matrix.
//   Stores qT/kT as [b][n][CQK] f16 (attention A/B frags contiguous),
//   v as [b][c][n] f16 (attention V B-frags contiguous).
// ---------------------------------------------------------------------------
__global__ __launch_bounds__(256) void qkv_project_kernel(
    const float* __restrict__ x,
    const _Float16* __restrict__ Wh,
    const float* __restrict__ bq, const float* __restrict__ bk,
    const float* __restrict__ bv,
    _Float16* __restrict__ qT, _Float16* __restrict__ kT,
    _Float16* __restrict__ v_ws)
{
    extern __shared__ _Float16 xs[];          // [NT1][CC] f16, transposed slab (64 KB)

    const int b  = blockIdx.y;
    const int n0 = blockIdx.x * NT1;
    const int t  = threadIdx.x;

    // Cooperative load of x slab [CC rows x NT1 cols] f32 -> LDS xs[n][c] f16.
    const float* xb = x + (size_t)b * CC * NN;
    for (int idx = t; idx < CC * NT1; idx += 256) {
        int c = idx >> 6;         // idx / NT1
        int n = idx & (NT1 - 1);  // idx % NT1
        xs[n * CC + c] = (_Float16)xb[(size_t)c * NN + n0 + n];
    }
    __syncthreads();

    const int wave = t >> 5;
    const int lane = t & 31;
    const int l16  = lane & 15;
    const int hi   = lane >> 4;            // 0 (lanes 0-15) / 1 (lanes 16-31)
    const int ns   = (wave & 3) * 16;      // wave's n-subtile in slab
    const int oBeg = (wave >> 2) * 320;    // wave's half of the 640 output rows

    for (int o0 = oBeg; o0 < oBeg + 320; o0 += 16) {
        // A-matrix rows come straight from f16 Wh: row o = o0 + l16 per lane.
        const _Float16* Wrow = Wh + (size_t)(o0 + l16) * CC;

        v8f acc = {};
        for (int c0 = 0; c0 < CC; c0 += 32) {
            // A frag: K = c0 + hi*8 + {0..7} and {16..23} (two 16B f16 loads)
            const _Float16* wp = Wrow + c0 + hi * 8;
            v16h Av = ld2x8h(wp, wp + 16);
            // B frag from LDS: col n = ns + l16, K = c0 + hi*8 + {0..7} and {16..23}
            const _Float16* xp = &xs[(ns + l16) * CC + c0 + hi * 8];
            v16h Bv = ld2x8h(xp, xp + 16);
            acc = __builtin_amdgcn_wmma_f32_16x16x32_f16(
                      false, Av, false, Bv, (short)0, acc, false, false);
        }

        // Bias: C/D element r maps to output row o = o0 + r + 8*hi.
        const int obase = o0 + hi * 8;
        #pragma unroll
        for (int r = 0; r < 8; ++r) {
            int oo = obase + r;
            float bb = (oo < CQK) ? bq[oo]
                     : (oo < 2 * CQK ? bk[oo - CQK] : bv[oo - 2 * CQK]);
            acc[r] += bb;
        }

        const int n = n0 + ns + l16;
        if (o0 < 2 * CQK) {
            // q/k: transposed store [b][n][olocal..olocal+7] -> one packed 16B store
            _Float16* dst   = (o0 < CQK) ? qT : kT;
            int olocal      = ((o0 < CQK) ? o0 : o0 - CQK) + hi * 8;
            union { v8h v; _Float16 e[8]; } pk;
            #pragma unroll
            for (int r = 0; r < 8; ++r) pk.e[r] = (_Float16)acc[r];
            *(v8h*)(dst + ((size_t)b * NN + n) * CQK + olocal) = pk.v;
        } else {
            // v: natural [b][c][n] layout (strided per-element stores)
            #pragma unroll
            for (int r = 0; r < 8; ++r) {
                int c = (o0 - 2 * CQK) + hi * 8 + r;
                v_ws[((size_t)b * CC + c) * NN + n] = (_Float16)acc[r];
            }
        }
    }
}

// ---------------------------------------------------------------------------
// Kernel 2: fused flash attention, 32 queries per workgroup.
//   8 waves x 64 channels = 512 output channels; 2 query sub-tiles of 16.
//   K tile (32x64 f16 = 4KB) double-buffered in LDS via async-to-LDS ops;
//   V B-fragments are reused by both query sub-tiles. Online softmax with
//   shfl-xor reductions; P relayout through wave-private LDS; O accumulated
//   in f32 WMMA registers.
// ---------------------------------------------------------------------------
__global__ __launch_bounds__(256) void attn_kernel(
    const float* __restrict__ x,
    const _Float16* __restrict__ qT,
    const _Float16* __restrict__ kT,
    const _Float16* __restrict__ v_ws,
    const float* __restrict__ gamma,
    float* __restrict__ out)
{
    __shared__ _Float16 Ktile[2][32][CQK];  // double-buffered K tile (8 KB)
    __shared__ _Float16 Plds[8][16][32];    // wave-private P tiles (8 KB)

    const int b    = blockIdx.y;
    const int i0   = blockIdx.x * QT;
    const int t    = threadIdx.x;
    const int wave = t >> 5;
    const int lane = t & 31;
    const int l16  = lane & 15;
    const int hi   = lane >> 4;
    const int cb   = wave * 64;             // wave's output-channel base

    // Cooperative K staging: 256 threads x 16B = one 32x64 f16 tile.
    const int krow   = t >> 3;              // 0..31
    const int kchunk = (t & 7) * 8;         // f16 offset within row
    const _Float16* kbase = kT + (size_t)b * NN * CQK;

    // Q A-fragments (2 query sub-tiles x 2 K-frags), loaded once.
    v16h Aq[2][2];
    #pragma unroll
    for (int it = 0; it < 2; ++it) {
        const _Float16* qrow = qT + ((size_t)b * NN + i0 + it * 16 + l16) * CQK;
        #pragma unroll
        for (int f = 0; f < 2; ++f) {
            const _Float16* p = qrow + f * 32 + hi * 8;
            Aq[it][f] = ld2x8h(p, p + 16);
        }
    }

    float m[2][8], lsum[2][8];
    #pragma unroll
    for (int it = 0; it < 2; ++it)
        #pragma unroll
        for (int r = 0; r < 8; ++r) { m[it][r] = -1e30f; lsum[it][r] = 0.f; }
    v8f O[2][4];
    #pragma unroll
    for (int it = 0; it < 2; ++it)
        #pragma unroll
        for (int ct = 0; ct < 4; ++ct) { v8f z = {}; O[it][ct] = z; }

    _Float16 (*P)[32] = Plds[wave];

    // Prologue: stage first K tile into buffer 0.
    stage16(&Ktile[0][krow][kchunk], kbase + (size_t)krow * CQK + kchunk);

    for (int j0 = 0; j0 < NN; j0 += 32) {
        const int buf  = (j0 >> 5) & 1;
        const bool has_next = (j0 + 32) < NN;
        if (has_next)
            stage16(&Ktile[buf ^ 1][krow][kchunk],
                    kbase + ((size_t)j0 + 32 + krow) * CQK + kchunk);
        if (has_next) { WAIT_ASYNC(1); } else { WAIT_ASYNC(0); }
        __syncthreads();   // K tile `buf` visible to all waves

        // Prefetch next V rows for this wave's channels.
        if (has_next)
            __builtin_prefetch(v_ws + ((size_t)b * CC + cb + l16) * NN + j0 + 32, 0, 3);

        v16h Ap[2];
        #pragma unroll
        for (int it = 0; it < 2; ++it) {
            // S = Q^T K over two 16-wide j-subtiles, K-dim = CQK (2 WMMA each).
            v8f S[2];
            #pragma unroll
            for (int tt = 0; tt < 2; ++tt) {
                v8f s = {};
                #pragma unroll
                for (int f = 0; f < 2; ++f) {
                    const _Float16* kp = &Ktile[buf][tt * 16 + l16][f * 32 + hi * 8];
                    v16h Bk = ld2x8h(kp, kp + 16);
                    s = __builtin_amdgcn_wmma_f32_16x16x32_f16(
                            false, Aq[it][f], false, Bk, (short)0, s, false, false);
                }
                S[tt] = s;
            }

            // Online softmax. Row i = r + 8*hi lives on the 16 lanes of this half.
            float scale[8];
            #pragma unroll
            for (int r = 0; r < 8; ++r) {
                float mx = fmaxf(S[0][r], S[1][r]);
                #pragma unroll
                for (int msk = 8; msk >= 1; msk >>= 1)
                    mx = fmaxf(mx, __shfl_xor(mx, msk, 32));
                float mn = fmaxf(m[it][r], mx);
                float sc = __expf(m[it][r] - mn);
                float p0 = __expf(S[0][r] - mn);
                float p1 = __expf(S[1][r] - mn);
                S[0][r] = p0; S[1][r] = p1;
                float rs = p0 + p1;
                #pragma unroll
                for (int msk = 8; msk >= 1; msk >>= 1)
                    rs += __shfl_xor(rs, msk, 32);
                lsum[it][r]  = lsum[it][r] * sc + rs;
                m[it][r]     = mn;
                scale[r]     = sc;
            }

            // Spill P (C/D layout) to wave-private LDS as row-major 16x32 f16.
            #pragma unroll
            for (int tt = 0; tt < 2; ++tt)
                #pragma unroll
                for (int r = 0; r < 8; ++r)
                    P[r + hi * 8][tt * 16 + l16] = (_Float16)S[tt][r];

            // Rescale accumulators by exp(m_old - m_new).
            #pragma unroll
            for (int ct = 0; ct < 4; ++ct)
                #pragma unroll
                for (int r = 0; r < 8; ++r)
                    O[it][ct][r] *= scale[r];

            // Reload P in A-frag layout: row i = l16, K(j) = hi*8 + {0..7,16..23}.
            const _Float16* pp = &P[l16][hi * 8];
            Ap[it] = ld2x8h(pp, pp + 16);
        }

        // O += P * V^T: each V B-fragment feeds both query sub-tiles.
        #pragma unroll
        for (int ct = 0; ct < 4; ++ct) {
            const _Float16* vp =
                v_ws + ((size_t)b * CC + cb + ct * 16 + l16) * NN + j0 + hi * 8;
            v16h Bv = ld2x8h(vp, vp + 16);
            O[0][ct] = __builtin_amdgcn_wmma_f32_16x16x32_f16(
                           false, Ap[0], false, Bv, (short)0, O[0][ct], false, false);
            O[1][ct] = __builtin_amdgcn_wmma_f32_16x16x32_f16(
                           false, Ap[1], false, Bv, (short)0, O[1][ct], false, false);
        }

        __syncthreads();   // all waves done with K tile `buf` before it is rewritten
    }

    // Epilogue: out = gamma * (O / l) + x
    const float g = gamma[0];
    #pragma unroll
    for (int it = 0; it < 2; ++it)
        #pragma unroll
        for (int ct = 0; ct < 4; ++ct) {
            int c = cb + ct * 16 + l16;
            #pragma unroll
            for (int r = 0; r < 8; ++r) {
                int i = i0 + it * 16 + r + hi * 8;
                size_t off = ((size_t)b * CC + c) * NN + i;
                out[off] = g * (O[it][ct][r] / lsum[it][r]) + x[off];
            }
        }
}

// ---------------------------------------------------------------------------
extern "C" void kernel_launch(void* const* d_in, const int* in_sizes, int n_in,
                              void* d_out, int out_size, void* d_ws, size_t ws_size,
                              hipStream_t stream) {
    const float* x     = (const float*)d_in[0];
    const float* Wq    = (const float*)d_in[1];
    const float* bq    = (const float*)d_in[2];
    const float* Wk    = (const float*)d_in[3];
    const float* bk    = (const float*)d_in[4];
    const float* Wv    = (const float*)d_in[5];
    const float* bv    = (const float*)d_in[6];
    const float* gamma = (const float*)d_in[7];
    float* out = (float*)d_out;

    // Workspace: qT (2 MB) | kT (2 MB) | v (16 MB) | Wh (640 KB), all f16.
    _Float16* qT   = (_Float16*)d_ws;
    _Float16* kT   = qT + (size_t)BB * NN * CQK;
    _Float16* v_ws = kT + (size_t)BB * NN * CQK;
    _Float16* Wh   = v_ws + (size_t)BB * CC * NN;

    dim3 blk(256);

    dim3 g0((OTOT * CC) / 256);
    convert_w_kernel<<<g0, blk, 0, stream>>>(Wq, Wk, Wv, Wh);

    dim3 g1(NN / NT1, BB);
    size_t lds1 = (size_t)CC * NT1 * sizeof(_Float16);   // 64 KB dynamic LDS
    qkv_project_kernel<<<g1, blk, lds1, stream>>>(x, Wh, bq, bk, bv,
                                                  qT, kT, v_ws);

    dim3 g2(NN / QT, BB);
    attn_kernel<<<g2, blk, 0, stream>>>(x, qT, kT, v_ws, gamma, out);
}
